// DaVinciMLP_7808250544928
// MI455X (gfx1250) — matrix operand order repaired and checked by reading, not compiled, not run on hardware
//
#include <hip/hip_runtime.h>
#include <hip/hip_bf16.h>

#define S_TOK  32768
#define DMODEL 2048
#define IDIM   8192
#define CHUNK  4096

typedef __attribute__((ext_vector_type(16))) __bf16 v16bf;
typedef __attribute__((ext_vector_type(8)))  float  v8f;

struct Pair16 { uint4 lo, hi; };

static __device__ inline v16bf load_frag(const uint4* lo, const uint4* hi) {
  Pair16 p; p.lo = *lo; p.hi = *hi;
  return __builtin_bit_cast(v16bf, p);
}

// ---------------------------------------------------------------------------
// Kernel 1: RMSNorm (fp32 stats on bf16-rounded input) + per-modality scale,
// output bf16. One 256-thread block (8 wave32) per row of 2048.
// ---------------------------------------------------------------------------
__global__ __launch_bounds__(256) void rmsnorm_kernel(
    const float* __restrict__ x,        // [CHUNK, D] (already chunk-offset)
    const float* __restrict__ norm_w,   // [E*D]
    __hip_bfloat16* __restrict__ xn,    // [CHUNK, D]
    const int* __restrict__ nvp, const int* __restrict__ nap,
    int rowOffset)
{
  const int lrow = blockIdx.x;
  const int grow = rowOffset + lrow;
  const int nv = *nvp, na = *nap;
  const int e = (grow >= nv + na) ? 2 : ((grow >= nv) ? 1 : 0);
  const int tid = threadIdx.x;

  const float* xr = x + (size_t)lrow * DMODEL;
  float vals[8];
  float ss = 0.f;
#pragma unroll
  for (int i = 0; i < 8; ++i) {
    int c = tid + i * 256;
    float v = __bfloat162float(__float2bfloat16(xr[c]));  // x -> bf16 -> fp32
    vals[i] = v;
    ss += v * v;
  }
  // wave32 reduction
#pragma unroll
  for (int o = 16; o > 0; o >>= 1) ss += __shfl_xor(ss, o, 32);
  __shared__ float red[8];
  if ((tid & 31) == 0) red[tid >> 5] = ss;
  __syncthreads();
  float tot = 0.f;
#pragma unroll
  for (int j = 0; j < 8; ++j) tot += red[j];
  const float inv = rsqrtf(tot * (1.0f / (float)DMODEL) + 1e-6f);

  const float* nw = norm_w + (size_t)e * DMODEL;
  __hip_bfloat16* orow = xn + (size_t)lrow * DMODEL;
#pragma unroll
  for (int i = 0; i < 8; ++i) {
    int c = tid + i * 256;
    orow[c] = __float2bfloat16(vals[i] * inv * (nw[c] + 1.0f));
  }
}

// ---------------------------------------------------------------------------
// Kernel 2/3: bf16 WMMA GEMM  C[M,N] = op(A[M,K] * B_e[N,K]^T)
//   B is [E*N, K] row-major; expert slab chosen from global row.
//   Block = 256x128 tile, 8 wave32 (4 M-waves x 2 N-waves), wave = 64x64
//   = 4x4 accumulators of v_wmma_f32_16x16x32_bf16 (128 acc VGPRs).
// A-frag per lane: 16B runs at K = 8*(lane/16) and +16 of row (m16 + lane%16)
// B-frag per lane: 16B runs at K = 8*(lane/16) and +16 of row (n16 + lane%16)
// ---------------------------------------------------------------------------
template <bool GELU>
__global__ __launch_bounds__(256) void gemm_bf16_wmma(
    const __hip_bfloat16* __restrict__ A,   // [Mchunk, K] bf16
    const __hip_bfloat16* __restrict__ Bw,  // [E*N, K]    bf16
    __hip_bfloat16* __restrict__ C,         // [Mchunk, N] bf16
    const int K, const int N,
    const int* __restrict__ nvp, const int* __restrict__ nap,
    const int rowOffset)
{
  const int tid  = threadIdx.x;
  const int lane = tid & 31;
  const int wv   = tid >> 5;
  const int wm   = wv & 3;        // 4 M-waves * 64 rows = 256
  const int wn   = wv >> 2;       // 2 N-waves * 64 cols = 128
  const int half = lane >> 4;     // lane group 0/1
  const int l16  = lane & 15;

  const int mBase = blockIdx.x * 256 + wm * 64;
  const int nBase = blockIdx.y * 128 + wn * 64;

  // expert from global row (split points are multiples of 256)
  const int grow = rowOffset + (int)blockIdx.x * 256;
  const int nv = *nvp, na = *nap;
  const int e = (grow >= nv + na) ? 2 : ((grow >= nv) ? 1 : 0);
  const __hip_bfloat16* Be = Bw + (size_t)e * (size_t)N * (size_t)K;

  // per-lane row pointers in uint4 (16B = 8 bf16) units
  const uint4* aRow[4];
  const uint4* bRow[4];
#pragma unroll
  for (int mt = 0; mt < 4; ++mt)
    aRow[mt] = (const uint4*)(A + (size_t)(mBase + mt * 16 + l16) * K) + half;
#pragma unroll
  for (int nt = 0; nt < 4; ++nt)
    bRow[nt] = (const uint4*)(Be + (size_t)(nBase + nt * 16 + l16) * K) + half;

  v8f acc[4][4];
#pragma unroll
  for (int mt = 0; mt < 4; ++mt)
#pragma unroll
    for (int nt = 0; nt < 4; ++nt)
      acc[mt][nt] = (v8f){0.f, 0.f, 0.f, 0.f, 0.f, 0.f, 0.f, 0.f};

  const int qsteps = K >> 3;            // K in uint4 units; k0+=32 -> q+=4
  for (int q = 0; q < qsteps; q += 4) {
    v16bf a[4], b[4];
#pragma unroll
    for (int nt = 0; nt < 4; ++nt)
      b[nt] = load_frag(bRow[nt] + q, bRow[nt] + q + 2);
#pragma unroll
    for (int mt = 0; mt < 4; ++mt)
      a[mt] = load_frag(aRow[mt] + q, aRow[mt] + q + 2);
#pragma unroll
    for (int mt = 0; mt < 4; ++mt)
#pragma unroll
      for (int nt = 0; nt < 4; ++nt)
        acc[mt][nt] = __builtin_amdgcn_wmma_f32_16x16x32_bf16(
            false, a[mt], false, b[nt], (short)0, acc[mt][nt], false, false);
  }

  // Epilogue: optional gelu7 (fast rcp sigmoid), convert to bf16, store.
  // v8f element r -> row = 8*half + r, col = l16 (16x16 f32 C/D layout).
#pragma unroll
  for (int mt = 0; mt < 4; ++mt) {
#pragma unroll
    for (int nt = 0; nt < 4; ++nt) {
      const int row = mBase + mt * 16 + half * 8;
      const int col = nBase + nt * 16 + l16;
      __hip_bfloat16* cp = C + (size_t)row * N + col;
#pragma unroll
      for (int r = 0; r < 8; ++r) {
        float v = acc[mt][nt][r];
        if (GELU) {
          v = fminf(v, 7.0f);
          // x * sigmoid(alpha*x); v_rcp_f32 instead of IEEE divide macro
          v = v * __builtin_amdgcn_rcpf(1.0f + __expf(-1.702f * v));
        }
        cp[(size_t)r * N] = __float2bfloat16(v);
      }
    }
  }
}

// ---------------------------------------------------------------------------
extern "C" void kernel_launch(void* const* d_in, const int* in_sizes, int n_in,
                              void* d_out, int out_size, void* d_ws, size_t ws_size,
                              hipStream_t stream) {
  const float*          x      = (const float*)d_in[0];
  const float*          norm_w = (const float*)d_in[1];
  const __hip_bfloat16* w_up   = (const __hip_bfloat16*)d_in[2];
  const __hip_bfloat16* w_down = (const __hip_bfloat16*)d_in[3];
  const int*            nv     = (const int*)d_in[4];
  const int*            na     = (const int*)d_in[5];
  __hip_bfloat16* out = (__hip_bfloat16*)d_out;

  __hip_bfloat16* xn  = (__hip_bfloat16*)d_ws;                    // [CHUNK, D]
  __hip_bfloat16* act = xn + (size_t)CHUNK * DMODEL;              // [CHUNK, I]

  for (int r0 = 0; r0 < S_TOK; r0 += CHUNK) {
    rmsnorm_kernel<<<CHUNK, 256, 0, stream>>>(
        x + (size_t)r0 * DMODEL, norm_w, xn, nv, na, r0);

    dim3 gUp(CHUNK / 256, IDIM / 128);
    gemm_bf16_wmma<true><<<gUp, 256, 0, stream>>>(
        xn, w_up, act, DMODEL, IDIM, nv, na, r0);

    dim3 gDn(CHUNK / 256, DMODEL / 128);
    gemm_bf16_wmma<false><<<gDn, 256, 0, stream>>>(
        act, w_down, out + (size_t)r0 * DMODEL, IDIM, DMODEL, nv, na, r0);
  }
}
